// RNNEnc_86930138071197
// MI455X (gfx1250) — matrix operand order repaired and checked by reading, not compiled, not run on hardware
//
#include <hip/hip_runtime.h>
#include <hip/hip_bf16.h>

#define DEV __device__ __forceinline__

typedef __bf16 bf16_t;
typedef bf16_t v16bf __attribute__((ext_vector_type(16)));
typedef float  v8f   __attribute__((ext_vector_type(8)));

// ---------------- problem sizes ----------------
constexpr int B    = 32;
constexpr int L    = 512;
constexpr int D    = 1024;
constexpr int G    = 3 * D;          // 3072 gate columns
constexpr int CTX  = 10;
constexpr int OUTT = L - 2 * CTX;    // 492
constexpr int ROWS = B * L;          // 16384 rows for the big GEMM

// recurrent kernel config
constexpr int RWG  = 64;             // workgroups per direction
constexpr int FPW  = 16;             // features per workgroup (one 16-wide N-tile per gate)
constexpr int COLS = 3 * FPW;        // 48 gate columns per WG
constexpr int LDK  = D + 8;          // LDS K-stride (bf16 elems): +16B pad -> conflict-free ds_load_b128
constexpr int WLDS_BYTES = COLS * LDK * 2;              // 99072 B of weights in LDS
constexpr int GH_BYTES   = B * COLS * 4;                // 6144 B gh tile
constexpr int SMEM_BYTES = WLDS_BYTES + GH_BYTES;       // 105216 B (fits 320KB/WGP LDS)

// ---------------- workspace layout (bytes) ----------------
constexpr size_t OFF_VBF = 0;                                  // [ROWS][D] bf16, row = t*32+b
constexpr size_t SZ_VBF  = (size_t)ROWS * D * 2;               // 32 MB
constexpr size_t OFF_WIH = OFF_VBF + SZ_VBF;                   // [2][G][D] bf16
constexpr size_t SZ_W    = (size_t)2 * G * D * 2;              // 12 MB
constexpr size_t OFF_WHH = OFF_WIH + SZ_W;                     // [2][G][D] bf16
constexpr size_t OFF_GI  = OFF_WHH + SZ_W;                     // [2][ROWS][G] f32 (b_ih folded in)
constexpr size_t SZ_GI   = (size_t)2 * ROWS * G * 4;           // 403 MB
constexpr size_t OFF_H   = OFF_GI + SZ_GI;                     // [2][B][D] f32
constexpr size_t SZ_H    = (size_t)2 * B * D * 4;
constexpr size_t OFF_HBF = OFF_H + SZ_H;                       // [2][2][B][D] bf16 (ping-pong)
constexpr size_t SZ_HBF  = (size_t)2 * 2 * B * D * 2;
constexpr size_t OFF_BAR = OFF_HBF + SZ_HBF;                   // [2][32] u32 barrier state

// ---------------- helpers ----------------
DEV v8f zero8() { v8f z = {0.f,0.f,0.f,0.f,0.f,0.f,0.f,0.f}; return z; }

union FragBF { v16bf v; uint4 q[2]; };

// A fragment (16x32 bf16, MxK): lanes 0-15 row M=lane%16, K k0..k0+15;
// lanes 16-31 same rows, K k0+16..k0+31.
DEV v16bf load_fragA(const bf16_t* __restrict__ base, int row, int k0, int lane, int ldk) {
  const int hlf = (lane >> 4) & 1;
  const bf16_t* p = base + (size_t)row * ldk + k0 + hlf * 16;
  FragBF f;
  f.q[0] = *(const uint4*)(p);
  f.q[1] = *(const uint4*)(p + 8);
  return f.v;
}

// B fragment (32x16 bf16, KxN), W row-major over K:
// lanes 0-15: col N=lane, K k0..k0+15 (32 contiguous bytes); lanes 16-31: K k0+16..k0+31.
DEV v16bf load_fragB(const bf16_t* __restrict__ wrow, int col, int k0, int lane, int ldk) {
  const int hlf = (lane >> 4) & 1;
  const bf16_t* p = wrow + (size_t)col * ldk + k0 + hlf * 16;
  FragBF f;
  f.q[0] = *(const uint4*)(p);
  f.q[1] = *(const uint4*)(p + 8);
  return f.v;
}

// B fragment from the padded LDS weight tile (ds_load_b128 x2, conflict-free)
DEV v16bf lds_fragB(const bf16_t* wlds, int ti, int lanecol, int k0, int hlf) {
  const bf16_t* p = wlds + (size_t)(ti * 16 + lanecol) * LDK + k0 + hlf * 16;
  FragBF f;
  f.q[0] = *(const uint4*)(p);
  f.q[1] = *(const uint4*)(p + 8);
  return f.v;
}

DEV v8f wmma_bf16(v16bf a, v16bf b, v8f c) {
  return __builtin_amdgcn_wmma_f32_16x16x32_bf16(false, a, false, b, (short)0, c, false, false);
}

DEV float fsigmoid(float x) { return 1.f / (1.f + __expf(-x)); }
DEV float ftanh(float x)    { float e = __expf(2.f * x); return 1.f - 2.f / (e + 1.f); }

// device-scope (per-direction) sense-reversal barrier: bar[0]=count, bar[1]=generation
DEV void grid_barrier(unsigned* bar, unsigned nwg) {
  __syncthreads();
  if (threadIdx.x == 0) {
    __threadfence();
    unsigned gen  = __hip_atomic_load(bar + 1, __ATOMIC_ACQUIRE, __HIP_MEMORY_SCOPE_AGENT);
    unsigned prev = __hip_atomic_fetch_add(bar, 1u, __ATOMIC_ACQ_REL, __HIP_MEMORY_SCOPE_AGENT);
    if (prev == nwg - 1u) {
      __hip_atomic_store(bar, 0u, __ATOMIC_RELAXED, __HIP_MEMORY_SCOPE_AGENT);
      __hip_atomic_fetch_add(bar + 1, 1u, __ATOMIC_RELEASE, __HIP_MEMORY_SCOPE_AGENT);
    } else {
      while (__hip_atomic_load(bar + 1, __ATOMIC_ACQUIRE, __HIP_MEMORY_SCOPE_AGENT) == gen) {
        __builtin_amdgcn_s_sleep(2);
      }
    }
  }
  __syncthreads();
}

// ---------------- prep kernels ----------------
__global__ void prep_v_kernel(const float* __restrict__ v, bf16_t* __restrict__ vbf) {
  const size_t n = (size_t)ROWS * D;
  for (size_t i = (size_t)blockIdx.x * blockDim.x + threadIdx.x; i < n;
       i += (size_t)gridDim.x * blockDim.x) {
    int d = (int)(i & (D - 1));
    int r = (int)(i >> 10);
    int t = r >> 5, b = r & 31;                 // row = t*32 + b
    vbf[i] = (bf16_t)v[((size_t)b * L + t) * D + d];
  }
}

__global__ void prep_w_kernel(const float* __restrict__ wih_f, const float* __restrict__ wih_b,
                              const float* __restrict__ whh_f, const float* __restrict__ whh_b,
                              bf16_t* __restrict__ wih, bf16_t* __restrict__ whh) {
  const float* src;
  bf16_t* dst;
  switch (blockIdx.y) {
    case 0:  src = wih_f; dst = wih;                     break;
    case 1:  src = wih_b; dst = wih + (size_t)G * D;     break;
    case 2:  src = whh_f; dst = whh;                     break;
    default: src = whh_b; dst = whh + (size_t)G * D;     break;
  }
  const size_t n = (size_t)G * D;
  for (size_t i = (size_t)blockIdx.x * blockDim.x + threadIdx.x; i < n;
       i += (size_t)gridDim.x * blockDim.x) {
    dst[i] = (bf16_t)src[i];
  }
}

__global__ void init_state_kernel(float* __restrict__ h, bf16_t* __restrict__ hbf,
                                  unsigned* __restrict__ bar) {
  const size_t nh = (size_t)2 * B * D;
  const size_t nb = (size_t)4 * B * D;
  for (size_t i = (size_t)blockIdx.x * blockDim.x + threadIdx.x; i < nh;
       i += (size_t)gridDim.x * blockDim.x) h[i] = 0.f;
  for (size_t i = (size_t)blockIdx.x * blockDim.x + threadIdx.x; i < nb;
       i += (size_t)gridDim.x * blockDim.x) hbf[i] = (bf16_t)0.f;
  if (blockIdx.x == 0 && threadIdx.x < 64) bar[threadIdx.x] = 0u;
}

// ---------------- phase 1: GI = x @ w_ih^T + b_ih (both directions) ----------------
// grid: (G/64, ROWS/256, 2), block 256 (8 waves). Wave owns a 32x64 tile: 2 M-tiles x 4 N-tiles
// => 8 WMMAs per 6 fragment loads (A reused over 4 N-tiles, B reused over 2 M-tiles).
__global__ __launch_bounds__(256) void gi_gemm_kernel(
    const bf16_t* __restrict__ vbf,     // [ROWS][D]
    const bf16_t* __restrict__ wih,     // [2][G][D]
    const float* __restrict__ bih_f, const float* __restrict__ bih_b,
    float* __restrict__ gi)             // [2][ROWS][G]
{
  const int lane = threadIdx.x & 31;
  const int wave = threadIdx.x >> 5;                 // 0..7
  const int dir  = blockIdx.z;
  const int rowBase = blockIdx.y * 256 + wave * 32;
  const int colBase = blockIdx.x * 64;
  const bf16_t* wp  = wih + (size_t)dir * G * D;
  const float* bias = dir ? bih_b : bih_f;

  v8f acc[2][4];
#pragma unroll
  for (int m = 0; m < 2; ++m)
#pragma unroll
    for (int n = 0; n < 4; ++n) acc[m][n] = zero8();

  const int arow0 = rowBase + (lane & 15);
  for (int k0 = 0; k0 < D; k0 += 32) {
    v16bf a0 = load_fragA(vbf, arow0,      k0, lane, D);
    v16bf a1 = load_fragA(vbf, arow0 + 16, k0, lane, D);
#pragma unroll
    for (int nt = 0; nt < 4; ++nt) {
      v16bf b = load_fragB(wp, colBase + nt * 16 + (lane & 15), k0, lane, D);
      acc[0][nt] = wmma_bf16(a0, b, acc[0][nt]);
      acc[1][nt] = wmma_bf16(a1, b, acc[1][nt]);
    }
  }

  const int hlf = (lane >> 4) & 1;
#pragma unroll
  for (int mt = 0; mt < 2; ++mt) {
#pragma unroll
    for (int nt = 0; nt < 4; ++nt) {
      int col = colBase + nt * 16 + (lane & 15);
      float bv = bias[col];
#pragma unroll
      for (int e = 0; e < 8; ++e) {
        int row = rowBase + mt * 16 + e + 8 * hlf;
        gi[((size_t)dir * ROWS + row) * G + col] = acc[mt][nt][e] + bv;
      }
    }
  }
}

// ---------------- phase 2: persistent recurrent kernel ----------------
// grid: (RWG, 2), block 64 (2 waves: wave = M-tile, each does the 3 gate N-tiles).
// Each WG owns 16 features; its w_hh slice (48 cols x 1024 K, bf16, 16B/col pad)
// is loaded into LDS ONCE and reused for all 512 steps (ds_load fragments).
__global__ __launch_bounds__(64) void gru_recurrent_kernel(
    const bf16_t* __restrict__ whh,   // [2][G][D]
    const float* __restrict__ gi,     // [2][ROWS][G] (b_ih folded in)
    const float* __restrict__ bhh_f, const float* __restrict__ bhh_b,
    const float* __restrict__ v_in,   // [B][L][D] fp32
    float* __restrict__ hst,          // [2][B][D] fp32 state
    bf16_t* __restrict__ hbf,         // [2][2][B][D] bf16 ping-pong
    unsigned* __restrict__ bar,       // [2][32]
    float* __restrict__ out)          // [B][OUTT][2D]
{
  extern __shared__ char smem[];
  bf16_t* wlds = (bf16_t*)smem;                      // [COLS][LDK] bf16, padded
  float*  gh   = (float*)(smem + WLDS_BYTES);        // [B][COLS] f32

  const int tid     = threadIdx.x;
  const int lane    = tid & 31;
  const int lanecol = lane & 15;
  const int mtile   = tid >> 5;       // wave id: 0..1 (batch halves)
  const int dir     = blockIdx.y;
  const int fbase   = blockIdx.x * FPW;
  const int hlf     = (lane >> 4) & 1;

  const bf16_t* wp  = whh + (size_t)dir * G * D;
  const float*  bhh = dir ? bhh_b : bhh_f;
  float*   h  = hst + (size_t)dir * B * D;
  bf16_t*  hb = hbf + (size_t)dir * 2 * B * D;
  unsigned* mybar = bar + dir * 32;

  // --- one-time: stage this WG's w_hh slice into LDS ---
  // local col c (0..47) -> global col (c/16)*1024 + fbase + (c%16); 16B chunks over K
  for (int idx = tid; idx < COLS * (D / 8); idx += 64) {
    int c  = idx >> 7;                // /128 chunks per column
    int kc = idx & 127;
    int gcol = (c >> 4) * D + fbase + (c & 15);
    uint4 v = *(const uint4*)(wp + (size_t)gcol * D + kc * 8);
    *(uint4*)(wlds + (size_t)c * LDK + kc * 8) = v;
  }
  __syncthreads();

  const int arow = mtile * 16 + lanecol;   // batch row for the A fragment

  for (int t = 0; t < L; ++t) {
    const bf16_t* hread  = hb + (size_t)(t & 1) * B * D;
    bf16_t*       hwrite = hb + (size_t)((t + 1) & 1) * B * D;

    // prefetch next step's GI slice into cache while we do this step's GEMM
    if (t + 1 < L) {
      const float* gn = gi + ((size_t)dir * ROWS + (size_t)(t + 1) * B) * G;
      for (int idx = tid; idx < B * 3; idx += 64) {
        int b = idx >> 2 >> 0; b = idx / 3; int gate = idx - b * 3;
        __builtin_prefetch(gn + (size_t)b * G + gate * D + fbase, 0, 1);
      }
    }

    // gh tile = h @ w_hh^T for this WG's 48 gate columns (weights from LDS)
    v8f acc0 = zero8(), acc1 = zero8(), acc2 = zero8();
    for (int k0 = 0; k0 < D; k0 += 32) {
      v16bf a  = load_fragA(hread, arow, k0, lane, D);
      v16bf b0 = lds_fragB(wlds, 0, lanecol, k0, hlf);
      acc0 = wmma_bf16(a, b0, acc0);
      v16bf b1 = lds_fragB(wlds, 1, lanecol, k0, hlf);
      acc1 = wmma_bf16(a, b1, acc1);
      v16bf b2 = lds_fragB(wlds, 2, lanecol, k0, hlf);
      acc2 = wmma_bf16(a, b2, acc2);
    }

#pragma unroll
    for (int i = 0; i < 3; ++i) {
      int lc = i * 16 + lanecol;                 // == gate*16 + fl
      v8f a = (i == 0) ? acc0 : ((i == 1) ? acc1 : acc2);
#pragma unroll
      for (int e = 0; e < 8; ++e) {
        int b = mtile * 16 + e + 8 * hlf;
        gh[b * COLS + lc] = a[e];
      }
    }
    __syncthreads();

    // gate nonlinearity + state update + fused output write (32 b x 16 f = 512 elems)
    const float* gip = gi + ((size_t)dir * ROWS + (size_t)t * B) * G;
    for (int idx = tid; idx < B * FPW; idx += 64) {
      int b  = idx >> 4;
      int fl = idx & 15;
      int f  = fbase + fl;
      const float* gb = gip + (size_t)b * G;
      float ghr = gh[b * COLS + fl]       + bhh[f];
      float ghz = gh[b * COLS + 16 + fl]  + bhh[D + f];
      float ghn = gh[b * COLS + 32 + fl]  + bhh[2 * D + f];
      float r = fsigmoid(gb[f]        + ghr);
      float z = fsigmoid(gb[D + f]    + ghz);
      float n = ftanh  (gb[2 * D + f] + r * ghn);
      float hprev = h[b * D + f];
      float hn = (1.f - z) * n + z * hprev;
      h[b * D + f]      = hn;
      hwrite[b * D + f] = (bf16_t)hn;
      if (t >= CTX && t < L - CTX) {
        int j  = t - CTX;
        int vt = dir ? (L - 1 - t) : t;
        float xv = v_in[((size_t)b * L + vt) * D + f];
        out[((size_t)b * OUTT + j) * (2 * D) + (size_t)dir * D + f] = hn + xv;
      }
    }

    grid_barrier(mybar, RWG);   // publish h(t) to all WGs of this direction
  }
}

// ---------------- host entry ----------------
extern "C" void kernel_launch(void* const* d_in, const int* in_sizes, int n_in,
                              void* d_out, int out_size, void* d_ws, size_t ws_size,
                              hipStream_t stream) {
  const float* v_in = (const float*)d_in[0];
  const float* wihf = (const float*)d_in[1];
  const float* whhf = (const float*)d_in[2];
  const float* bihf = (const float*)d_in[3];
  const float* bhhf = (const float*)d_in[4];
  const float* wihb = (const float*)d_in[5];
  const float* whhb = (const float*)d_in[6];
  const float* bihb = (const float*)d_in[7];
  const float* bhhb = (const float*)d_in[8];

  char* ws = (char*)d_ws;
  bf16_t*  vbf = (bf16_t*)(ws + OFF_VBF);
  bf16_t*  wih = (bf16_t*)(ws + OFF_WIH);
  bf16_t*  whh = (bf16_t*)(ws + OFF_WHH);
  float*   gi  = (float*)(ws + OFF_GI);
  float*   h   = (float*)(ws + OFF_H);
  bf16_t*  hbf = (bf16_t*)(ws + OFF_HBF);
  unsigned* bar = (unsigned*)(ws + OFF_BAR);
  float*   out = (float*)d_out;

  // allow >64KB dynamic LDS for the persistent kernel (CDNA5: 320KB per WG)
  (void)hipFuncSetAttribute((const void*)gru_recurrent_kernel,
                            hipFuncAttributeMaxDynamicSharedMemorySize, SMEM_BYTES);

  prep_v_kernel<<<2048, 256, 0, stream>>>(v_in, vbf);
  prep_w_kernel<<<dim3(1024, 4), 256, 0, stream>>>(wihf, wihb, whhf, whhb, wih, whh);
  init_state_kernel<<<64, 256, 0, stream>>>(h, hbf, bar);

  gi_gemm_kernel<<<dim3(G / 64, ROWS / 256, 2), 256, 0, stream>>>(vbf, wih, bihf, bihb, gi);

  gru_recurrent_kernel<<<dim3(RWG, 2), 64, SMEM_BYTES, stream>>>(whh, gi, bhhf, bhhb, v_in,
                                                                 h, hbf, bar, out);
}
